// PoseReIDModel_82377472737927
// MI455X (gfx1250) — compile-verified, matrix operand order checked by reading, not run on hardware
//
#include <hip/hip_runtime.h>
#include <hip/hip_bf16.h>
#include <math.h>

typedef _Float16 v4h  __attribute__((ext_vector_type(4)));
typedef _Float16 v8h  __attribute__((ext_vector_type(8)));
typedef _Float16 v16h __attribute__((ext_vector_type(16)));
typedef float    v8f  __attribute__((ext_vector_type(8)));

#define NN 64
#define CC 2048
#define HH 24
#define WW 12
#define HWN 288
#define PP 5
#define CL 256
#define CM 3328   // 2048 + 5*256
#define BN_EPS 1e-5f
#define LDSPITCH 136   // 128 + 8 half pad -> conflict-free ds_load_b128

// ---------------- box bounds: one thread per (n,p) ----------------
__global__ void box_kernel(const float* __restrict__ poses, int* __restrict__ boxes) {
    int t = threadIdx.x;
    if (t >= NN * PP) return;
    int n = t / PP, p = t % PP;
    const float* kp = poses + ((size_t)n * 17 + p * 4) * 4;   // POSE_IDS = 4*p
    float kx = kp[0], ky = kp[1], conf = kp[2];
    float bx = fminf(fmaxf(kx - 0.25f, 0.f), 0.75f) * conf;
    float by = fminf(fmaxf(ky - 0.25f, 0.f), 0.75f) * conf;
    float bw = 0.5f * conf;   // clip(0.5,0,0.75)=0.5
    float bh = 0.5f * conf;
    int xs = max(0, (int)rintf((float)WW * bx));
    int xe = min(WW - 1, (int)rintf((float)WW * (bx + bw)));
    int ys = max(0, (int)rintf((float)HH * by));
    int ye = min(HH - 1, (int)rintf((float)HH * (by + bh)));
    int area = (ye - ys) * (xe - xs);
    int* o = boxes + t * 5;
    o[0] = xs; o[1] = xe; o[2] = ys; o[3] = ye; o[4] = area;
}

// ---- fused single pass over x: 5 masked averages + global max ----
// one wave32 per (n,c); 288 contiguous floats, lane-strided (coalesced).
__global__ void pool_kernel(const float* __restrict__ x, const int* __restrict__ boxes,
                            _Float16* __restrict__ avg_h,   // [P][N][C] f16
                            _Float16* __restrict__ cat) {   // [N][CM]  f16, gfeat at +1280
    int wave = (blockIdx.x * blockDim.x + threadIdx.x) >> 5;
    int lane = threadIdx.x & 31;
    int n = wave >> 11;        // / 2048
    int c = wave & 2047;
    const float* xr = x + ((size_t)n * CC + c) * HWN;
    const int* bb = boxes + n * PP * 5;
    int xs[PP], xe[PP], ys[PP], ye[PP];
    float inv[PP];
    #pragma unroll
    for (int p = 0; p < PP; ++p) {
        xs[p] = bb[p * 5 + 0]; xe[p] = bb[p * 5 + 1];
        ys[p] = bb[p * 5 + 2]; ye[p] = bb[p * 5 + 3];
        int a = bb[p * 5 + 4];
        inv[p] = 1.0f / (float)(a > 0 ? a : 1);
    }
    float mx = -INFINITY;
    float s[PP] = {0.f, 0.f, 0.f, 0.f, 0.f};
    #pragma unroll
    for (int i = 0; i < 9; ++i) {            // 288 = 9 * 32
        int idx = lane + i * 32;
        float v = xr[idx];
        int h = idx / WW, w = idx - h * WW;
        mx = fmaxf(mx, v);
        #pragma unroll
        for (int p = 0; p < PP; ++p) {
            bool in = (h >= ys[p]) && (h < ye[p]) && (w >= xs[p]) && (w < xe[p]);
            s[p] += in ? v : 0.f;
        }
    }
    #pragma unroll
    for (int off = 16; off >= 1; off >>= 1) {
        mx = fmaxf(mx, __shfl_xor(mx, off, 32));
        #pragma unroll
        for (int p = 0; p < PP; ++p) s[p] += __shfl_xor(s[p], off, 32);
    }
    if (lane == 0) {
        cat[(size_t)n * CM + 1280 + c] = (_Float16)mx;
        #pragma unroll
        for (int p = 0; p < PP; ++p)
            avg_h[((size_t)p * NN + n) * CC + c] = (_Float16)(s[p] * inv[p]);
    }
}

// ---------------- WMMA helpers ----------------
__device__ inline v16h cat16(v8h lo, v8h hi) {
    return __builtin_shufflevector(lo, hi, 0,1,2,3,4,5,6,7,8,9,10,11,12,13,14,15);
}

// GEMM core: block = 128 threads (4 waves). Wave w owns the 16-row M-tile
// [16w, 16w+16) of a 16-column output tile; the 16 B rows (K-contiguous f32)
// are staged once per block through LDS as f16 (shared by all 4 waves), in
// K chunks of 128. A is read directly (f16, K-contiguous).
__device__ inline v8f gemm_core(const _Float16* __restrict__ A, int ldA,
                                const float* __restrict__ Brows, int ldB,
                                int K, _Float16* bsh) {
    int tid  = threadIdx.x;
    int lane = tid & 31;
    int w    = tid >> 5;          // wave = M-tile index 0..3
    int row  = lane & 15;
    int sel  = lane >> 4;
    // cooperative B-chunk load assignment: 16 rows x 8 segments of 16 floats
    int brow = tid >> 3;          // 0..15
    int bseg = tid & 7;           // 0..7
    const float*    bsrc = Brows + (size_t)brow * ldB + bseg * 16;
    _Float16*       bdst = bsh + brow * LDSPITCH + bseg * 16;
    const _Float16* a0   = A + (size_t)(w * 16 + row) * ldA;

    v8f acc = {};
    for (int k0 = 0; k0 < K; k0 += 128) {
        if (k0 + 128 < K) __builtin_prefetch(bsrc + k0 + 128, 0, 3);  // global_prefetch_b8
        #pragma unroll
        for (int q = 0; q < 4; ++q) {       // 4 x float4 = 16 floats/thread
            float4 f = *(const float4*)(bsrc + k0 + q * 4);
            v4h h = { (_Float16)f.x, (_Float16)f.y, (_Float16)f.z, (_Float16)f.w };
            *(v4h*)(bdst + q * 4) = h;
        }
        __syncthreads();
        #pragma unroll
        for (int kk = 0; kk < 4; ++kk) {    // 4 WMMA steps of K=32
            int kb = kk * 32;
            v16h a = cat16(*(const v8h*)(a0 + k0 + kb + sel * 8),
                           *(const v8h*)(a0 + k0 + kb + 16 + sel * 8));
            const _Float16* bp = bsh + row * LDSPITCH + kb;
            v16h b = cat16(*(const v8h*)(bp + sel * 8),
                           *(const v8h*)(bp + 16 + sel * 8));
            acc = __builtin_amdgcn_wmma_f32_16x16x32_f16(false, a, false, b,
                                                         (short)0, acc, false, false);
        }
        __syncthreads();
    }
    return acc;
}

// ---- GEMM1: loc[n,p,k] = avg[n,p,:] . local_w[p,k,:]; fused bias+BN+ReLU -> cat f16 ----
// grid = (CL/16, P), block = 128 (all of M=64 per block; weights read once)
__global__ __launch_bounds__(128) void gemm1_kernel(
        const _Float16* __restrict__ avg_h, const float* __restrict__ lw,
        const float* __restrict__ lb, const float* __restrict__ lg,
        const float* __restrict__ lbeta, const float* __restrict__ lm,
        const float* __restrict__ lv, _Float16* __restrict__ cat) {
    __shared__ __align__(16) _Float16 bsh[16 * LDSPITCH];
    int p = blockIdx.y;
    int colBase = blockIdx.x * 16;
    const _Float16* A = avg_h + (size_t)p * NN * CC;
    const float*    B = lw + ((size_t)p * CL + colBase) * CC;
    v8f acc = gemm_core(A, CC, B, CC, CC, bsh);

    int lane = threadIdx.x & 31, w = threadIdx.x >> 5;
    int row = lane & 15, sel = lane >> 4;
    int kflat = p * CL + colBase + row;          // column in cat / param index
    float bias  = lb[kflat];
    float scale = lg[kflat] * rsqrtf(lv[kflat] + BN_EPS);
    float mean  = lm[kflat];
    float beta  = lbeta[kflat];
    #pragma unroll
    for (int r = 0; r < 8; ++r) {
        int m = w * 16 + r + sel * 8;
        float v = (acc[r] + bias - mean) * scale + beta;
        v = fmaxf(v, 0.f);
        cat[(size_t)m * CM + kflat] = (_Float16)v;
    }
}

// ---- GEMM2: y[n,j] = cat[n,:] . merge_w[j,:]; fused bias+BN+ReLU -> out f32 ----
// grid = (C/16), block = 128 (all of M=64 per block; merge_w read once)
__global__ __launch_bounds__(128) void gemm2_kernel(
        const _Float16* __restrict__ cat, const float* __restrict__ mw,
        const float* __restrict__ mb, const float* __restrict__ mg,
        const float* __restrict__ mbeta, const float* __restrict__ mm,
        const float* __restrict__ mv, float* __restrict__ y) {
    __shared__ __align__(16) _Float16 bsh[16 * LDSPITCH];
    int colBase = blockIdx.x * 16;
    const float* B = mw + (size_t)colBase * CM;
    v8f acc = gemm_core(cat, CM, B, CM, CM, bsh);

    int lane = threadIdx.x & 31, w = threadIdx.x >> 5;
    int row = lane & 15, sel = lane >> 4;
    int j = colBase + row;
    float bias  = mb[j];
    float scale = mg[j] * rsqrtf(mv[j] + BN_EPS);
    float mean  = mm[j];
    float beta  = mbeta[j];
    #pragma unroll
    for (int r = 0; r < 8; ++r) {
        int m = w * 16 + r + sel * 8;
        float v = (acc[r] + bias - mean) * scale + beta;
        y[(size_t)m * CC + j] = fmaxf(v, 0.f);
    }
}

// ---- in-place row L2 normalize: block per sample ----
__global__ void norm_kernel(float* __restrict__ y) {
    __shared__ float red[256];
    int n = blockIdx.x, t = threadIdx.x;
    float* yr = y + (size_t)n * CC;
    float s = 0.f;
    for (int j = t; j < CC; j += 256) { float v = yr[j]; s += v * v; }
    red[t] = s;
    __syncthreads();
    for (int st = 128; st > 0; st >>= 1) {
        if (t < st) red[t] += red[t + st];
        __syncthreads();
    }
    float scale = 1.0f / fmaxf(sqrtf(red[0]), 1e-12f);
    for (int j = t; j < CC; j += 256) yr[j] *= scale;   // each thread owns its j's
}

extern "C" void kernel_launch(void* const* d_in, const int* in_sizes, int n_in,
                              void* d_out, int out_size, void* d_ws, size_t ws_size,
                              hipStream_t stream) {
    const float* x     = (const float*)d_in[0];
    const float* poses = (const float*)d_in[1];
    const float* lw    = (const float*)d_in[2];
    const float* lb    = (const float*)d_in[3];
    const float* lg    = (const float*)d_in[4];
    const float* lbeta = (const float*)d_in[5];
    const float* lm    = (const float*)d_in[6];
    const float* lv    = (const float*)d_in[7];
    const float* mw    = (const float*)d_in[8];
    const float* mb    = (const float*)d_in[9];
    const float* mg    = (const float*)d_in[10];
    const float* mbeta = (const float*)d_in[11];
    const float* mm    = (const float*)d_in[12];
    const float* mv    = (const float*)d_in[13];
    float* out = (float*)d_out;

    char* ws = (char*)d_ws;
    int*      boxes = (int*)ws;                               // 320*5*4 = 6.4 KB
    size_t off = 8192;
    _Float16* avg_h = (_Float16*)(ws + off);                  // P*N*C f16 = 1.31 MB
    off += (size_t)PP * NN * CC * sizeof(_Float16);
    _Float16* cat   = (_Float16*)(ws + off);                  // N*CM f16 = 0.43 MB
    // total ws use ~1.75 MB; y lives in d_out (normalized in-place).

    box_kernel<<<1, NN * PP, 0, stream>>>(poses, boxes);

    // one wave32 per (n,c): N*C/8 blocks of 256 threads
    pool_kernel<<<(NN * CC) / 8, 256, 0, stream>>>(x, boxes, avg_h, cat);

    gemm1_kernel<<<dim3(CL / 16, PP), 128, 0, stream>>>(
        avg_h, lw, lb, lg, lbeta, lm, lv, cat);

    gemm2_kernel<<<dim3(CC / 16), 128, 0, stream>>>(
        cat, mw, mb, mg, mbeta, mm, mv, out);

    norm_kernel<<<NN, 256, 0, stream>>>(out);
}
